// Graph_Transformer_67010079752647
// MI455X (gfx1250) — matrix operand order repaired and checked by reading, not compiled, not run on hardware
//
#include <hip/hip_runtime.h>
#include <hip/hip_bf16.h>
#include <stdint.h>

#define NN   50000
#define NE   400000
#define NH   8
#define INC  128
#define D1   256        /* H * HID_C = H * OUT_C */
#define OC   32
#define CH   32         /* channels per head */
#define RSQRT_CH 0.17677669529663689f
#define MTILES (NN / 16)   /* 3125 */

typedef __attribute__((ext_vector_type(16))) _Float16 v16h;
typedef __attribute__((ext_vector_type(8)))  _Float16 v8h;
typedef __attribute__((ext_vector_type(8)))  float    v8f;
typedef long long ll;

// ---------------- elementwise helpers ----------------

__global__ void fill_u32_kernel(uint32_t* __restrict__ p, uint32_t v, int n) {
    int i = blockIdx.x * blockDim.x + threadIdx.x;
    if (i < n) p[i] = v;
}

__global__ void cast_f32_f16_kernel(const float* __restrict__ in,
                                    _Float16* __restrict__ out, int n) {
    int i = blockIdx.x * blockDim.x + threadIdx.x;
    if (i < n) out[i] = (_Float16)in[i];
}

// h = act(agg + skip), cast to f16 for the next layer's WMMA GEMMs
__global__ void add_act_cast_kernel(const float* __restrict__ a,
                                    const float* __restrict__ b,
                                    _Float16* __restrict__ out, int n, int relu) {
    int i = blockIdx.x * blockDim.x + threadIdx.x;
    if (i < n) {
        float s = a[i] + b[i];
        if (relu) s = fmaxf(s, 0.0f);
        out[i] = (_Float16)s;
    }
}

// ---------------- weight packing into WMMA B-fragment order ----------------
// B fragment (16-bit, 32x16 KxN), contiguous k order chosen here (A matches): lane L holds n = L%16, K = (L/16)*16 + j, j=0..15
// Packed layout: Bp[((kt*nColTiles + nt)*32 + lane)*16 + j]
__global__ void pack_weight_kernel(const float* __restrict__ W,
                                   _Float16* __restrict__ Bp, int Nc, int total) {
    int idx = blockIdx.x * blockDim.x + threadIdx.x;
    if (idx >= total) return;
    int j    = idx & 15;
    int lane = (idx >> 4) & 31;
    int t    = idx >> 9;                 // kt*nColTiles + nt
    int nColTiles = Nc >> 4;
    int nt = t % nColTiles;
    int kt = t / nColTiles;
    int n  = (nt << 4) + (lane & 15);
    int k  = (kt << 5) + ((lane >> 4) << 4) + j;
    Bp[idx] = (_Float16)W[(size_t)k * Nc + n];
}

// ---------------- WMMA GEMM with async-staged LDS weight panel ----------------
// Block = 256 threads (8 waves). Each wave owns TWO 16-row tiles; all waves
// share the same NT*16-wide column group whose packed weight panel
// (kTiles x NT x 1KB) is staged global->LDS once per block via
// global_load_async_to_lds_b128 (ASYNCcnt), then read with ds_load_b128.
// Per k-step: 2 A fragments (global), NT B fragments (LDS), 2*NT WMMAs.
template <int NT>
__global__ void __launch_bounds__(256)
gemm_wmma_lds_kernel(const _Float16* __restrict__ A,
                     const _Float16* __restrict__ Bp,
                     const float*    __restrict__ bias,
                     float*          __restrict__ C,
                     int K, int Nc, int mTiles) {
    __shared__ __align__(16) _Float16 sB[8 * NT * 512];  // up to 8 k-steps

    const int tid  = threadIdx.x;
    const int lane = tid & 31;
    const int wave = tid >> 5;
    const int half = lane >> 4;
    const int l16  = lane & 15;
    const int ct0  = blockIdx.y * NT;
    const int nColTiles = Nc >> 4;
    const int kTiles    = K >> 5;

    // ---- async stage of the B panel: kTiles * NT * 1024 bytes ----
    {
        const int chunks = kTiles * NT * 64;          // 16-byte chunks
        for (int c = tid; c < chunks; c += 256) {
            int kt  = c / (NT * 64);
            int rem = c - kt * (NT * 64);             // chunk within k-step
            uint32_t ldsoff = (uint32_t)(uintptr_t)(&sB[kt * NT * 512 + rem * 8]);
            const _Float16* g = Bp + ((size_t)(kt * nColTiles + ct0) * 512 + rem * 8);
            asm volatile("global_load_async_to_lds_b128 %0, %1, off"
                         :: "v"(ldsoff), "v"(g) : "memory");
        }
        asm volatile("s_wait_asynccnt 0x0" ::: "memory");
    }
    __syncthreads();

    const int r0 = (blockIdx.x * 8 + wave) * 2;   // first of two row tiles
    const int r1 = r0 + 1;

    v8f acc0[NT], acc1[NT];
    const v8f vzero = {0.f, 0.f, 0.f, 0.f, 0.f, 0.f, 0.f, 0.f};
#pragma unroll
    for (int t = 0; t < NT; ++t) { acc0[t] = vzero; acc1[t] = vzero; }

    // A-fragment 16-bit A 16x32 layout (contiguous, matches packed B):
    //   half==0: elems 0..15 = K 0..15
    //   half==1: elems 0..15 = K 16..31
    const _Float16* Arow0 = A + (size_t)(r0 * 16 + l16) * K + half * 16;
    const _Float16* Arow1 = Arow0 + (size_t)16 * K;

    for (int kt = 0; kt < kTiles; ++kt) {
        v8h a0lo = *(const v8h*)(Arow0);
        v8h a0hi = *(const v8h*)(Arow0 + 8);
        v8h a1lo = *(const v8h*)(Arow1);
        v8h a1hi = *(const v8h*)(Arow1 + 8);
        Arow0 += 32; Arow1 += 32;
        v16h a0 = __builtin_shufflevector(a0lo, a0hi,
                                          0,1,2,3,4,5,6,7,8,9,10,11,12,13,14,15);
        v16h a1 = __builtin_shufflevector(a1lo, a1hi,
                                          0,1,2,3,4,5,6,7,8,9,10,11,12,13,14,15);
        const _Float16* bb = &sB[kt * NT * 512 + lane * 16];
        v16h bfrag[NT];
#pragma unroll
        for (int t = 0; t < NT; ++t)
            bfrag[t] = *(const v16h*)(bb + t * 512);   // all ds_loads first
#pragma unroll
        for (int t = 0; t < NT; ++t)
            acc0[t] = __builtin_amdgcn_wmma_f32_16x16x32_f16(
                false, a0, false, bfrag[t], (short)0, acc0[t], false, false);
#pragma unroll
        for (int t = 0; t < NT; ++t)
            acc1[t] = __builtin_amdgcn_wmma_f32_16x16x32_f16(
                false, a1, false, bfrag[t], (short)0, acc1[t], false, false);
    }

    // D layout: VGPR r, lane L -> row = 8*half + r, col = L%16
    if (r0 < mTiles) {
        const int rbase = r0 * 16 + half * 8;
#pragma unroll
        for (int t = 0; t < NT; ++t) {
            int n = (ct0 + t) * 16 + l16;
            float bv = bias[n];
#pragma unroll
            for (int r = 0; r < 8; ++r)
                C[(size_t)(rbase + r) * Nc + n] = acc0[t][r] + bv;
        }
    }
    if (r1 < mTiles) {
        const int rbase = r1 * 16 + half * 8;
#pragma unroll
        for (int t = 0; t < NT; ++t) {
            int n = (ct0 + t) * 16 + l16;
            float bv = bias[n];
#pragma unroll
            for (int r = 0; r < 8; ++r)
                C[(size_t)(rbase + r) * Nc + n] = acc1[t][r] + bv;
        }
    }
}

// ---------------- attention (edge) kernels ----------------

__device__ __forceinline__ void atomicMaxFloat(float* addr, float val) {
    if (val >= 0.0f) atomicMax((int*)addr, __float_as_int(val));
    else             atomicMin((unsigned int*)addr, __float_as_uint(val));
}

// logits[e,h] = <q[dst], k[src]> / sqrt(CH);  also running segment-max per (dst,h)
__global__ void edge_logits_kernel(const float* __restrict__ q,
                                   const float* __restrict__ k,
                                   const ll* __restrict__ src,
                                   const ll* __restrict__ dst,
                                   float* __restrict__ logits,
                                   float* __restrict__ nodeMax) {
    int idx = blockIdx.x * blockDim.x + threadIdx.x;
    if (idx >= NE * NH) return;
    int e = idx >> 3, h = idx & 7;
    ll s = src[e], d = dst[e];
    const float4* qp = (const float4*)(q + (size_t)d * D1 + h * CH);
    const float4* kp = (const float4*)(k + (size_t)s * D1 + h * CH);
    float acc = 0.f;
#pragma unroll
    for (int i = 0; i < 8; ++i) {
        float4 a = qp[i], b = kp[i];
        acc += a.x * b.x + a.y * b.y + a.z * b.z + a.w * b.w;
    }
    float lg = acc * RSQRT_CH;
    logits[idx] = lg;
    atomicMaxFloat(&nodeMax[(size_t)d * NH + h], lg);
}

// e = exp(logit - max[dst]); denom[dst,h] += e  (in-place over logits)
__global__ void edge_exp_kernel(float* __restrict__ logits,
                                const float* __restrict__ nodeMax,
                                float* __restrict__ denom,
                                const ll* __restrict__ dst) {
    int idx = blockIdx.x * blockDim.x + threadIdx.x;
    if (idx >= NE * NH) return;
    int e = idx >> 3, h = idx & 7;
    ll d = dst[e];
    float ev = __expf(logits[idx] - nodeMax[(size_t)d * NH + h]);
    logits[idx] = ev;
    atomicAdd(&denom[(size_t)d * NH + h], ev);
}

// agg[dst,h,:] += (e/denom[dst,h]) * v[src,h,:]
__global__ void edge_aggregate_kernel(const float* __restrict__ eexp,
                                      const float* __restrict__ denom,
                                      const float* __restrict__ v,
                                      const ll* __restrict__ src,
                                      const ll* __restrict__ dst,
                                      float* __restrict__ agg) {
    int idx = blockIdx.x * blockDim.x + threadIdx.x;
    if (idx >= NE * NH) return;
    int e = idx >> 3, h = idx & 7;
    ll s = src[e], d = dst[e];
    float alpha = eexp[idx] / denom[(size_t)d * NH + h];
    const float4* vp = (const float4*)(v + (size_t)s * D1 + h * CH);
    float* ap = agg + (size_t)d * D1 + h * CH;
#pragma unroll
    for (int i = 0; i < 8; ++i) {
        float4 vv = vp[i];
        atomicAdd(ap + i * 4 + 0, alpha * vv.x);
        atomicAdd(ap + i * 4 + 1, alpha * vv.y);
        atomicAdd(ap + i * 4 + 2, alpha * vv.z);
        atomicAdd(ap + i * 4 + 3, alpha * vv.w);
    }
}

// ---------------- final mean over nodes ----------------
__global__ void mean_reduce_kernel(const float* __restrict__ tmp,
                                   float* __restrict__ out) {
    int tid = blockIdx.x * blockDim.x + threadIdx.x;
    int c = tid & 31;
    int row = tid >> 5;
    int stride = (gridDim.x * blockDim.x) >> 5;
    float s = 0.f;
    for (int n = row; n < NN; n += stride)
        s += tmp[(size_t)n * OC + c];
    atomicAdd(&out[c], s * (1.0f / (float)NN));
}

// ---------------- launcher ----------------

static inline void* bump(char*& p, size_t bytes) {
    void* r = (void*)p;
    p += (bytes + 255) & ~(size_t)255;
    return r;
}

extern "C" void kernel_launch(void* const* d_in, const int* in_sizes, int n_in,
                              void* d_out, int out_size, void* d_ws, size_t ws_size,
                              hipStream_t stream) {
    (void)in_sizes; (void)n_in; (void)out_size; (void)ws_size;

    const float* x    = (const float*)d_in[0];
    const ll*    eidx = (const ll*)d_in[1];
    const ll*    src  = eidx;
    const ll*    dst  = eidx + NE;
    const float* Wl1[4] = {(const float*)d_in[2], (const float*)d_in[4],
                           (const float*)d_in[6], (const float*)d_in[8]};   // q1,k1,v1,s1
    const float* Bl1[4] = {(const float*)d_in[3], (const float*)d_in[5],
                           (const float*)d_in[7], (const float*)d_in[9]};
    const float* Wl2[4] = {(const float*)d_in[10], (const float*)d_in[12],
                           (const float*)d_in[14], (const float*)d_in[16]}; // q2,k2,v2,s2
    const float* Bl2[4] = {(const float*)d_in[11], (const float*)d_in[13],
                           (const float*)d_in[15], (const float*)d_in[17]};
    const float* linW = (const float*)d_in[18];
    const float* linB = (const float*)d_in[19];
    float* out = (float*)d_out;

    // ---- workspace layout ----
    char* p = (char*)d_ws;
    _Float16* wp1[4]; for (int i = 0; i < 4; ++i) wp1[i] = (_Float16*)bump(p, (size_t)INC * D1 * 2);
    _Float16* wp2[4]; for (int i = 0; i < 4; ++i) wp2[i] = (_Float16*)bump(p, (size_t)D1 * D1 * 2);
    _Float16* wplin = (_Float16*)bump(p, (size_t)D1 * OC * 2);
    _Float16* xh  = (_Float16*)bump(p, (size_t)NN * INC * 2);
    _Float16* hH  = (_Float16*)bump(p, (size_t)NN * D1 * 2);   // h1 (f16), reused for h2
    float* qf   = (float*)bump(p, (size_t)NN * D1 * 4);
    float* kf   = (float*)bump(p, (size_t)NN * D1 * 4);
    float* vf   = (float*)bump(p, (size_t)NN * D1 * 4);
    float* sf   = (float*)bump(p, (size_t)NN * D1 * 4);
    float* agg  = (float*)bump(p, (size_t)NN * D1 * 4);
    float* eexp = (float*)bump(p, (size_t)NE * NH * 4);
    float* nmax = (float*)bump(p, (size_t)NN * NH * 4);
    float* den  = (float*)bump(p, (size_t)NN * NH * 4);
    float* tmpo = (float*)bump(p, (size_t)NN * OC * 4);

    const int TPB = 256;
    auto blocks = [](size_t n, int b) { return (unsigned)((n + b - 1) / b); };

    // ---- pack weights to WMMA fragment order (tiny, once per launch) ----
    for (int i = 0; i < 4; ++i) {
        int tot = INC * D1;
        pack_weight_kernel<<<blocks(tot, TPB), TPB, 0, stream>>>(Wl1[i], wp1[i], D1, tot);
    }
    for (int i = 0; i < 4; ++i) {
        int tot = D1 * D1;
        pack_weight_kernel<<<blocks(tot, TPB), TPB, 0, stream>>>(Wl2[i], wp2[i], D1, tot);
    }
    {
        int tot = D1 * OC;
        pack_weight_kernel<<<blocks(tot, TPB), TPB, 0, stream>>>(linW, wplin, OC, tot);
    }

    // ---- layer 1 ----
    cast_f32_f16_kernel<<<blocks((size_t)NN * INC, TPB), TPB, 0, stream>>>(x, xh, NN * INC);

    // 8 waves/block, 2 row-tiles/wave -> 16 row tiles per block
    dim3 g1((MTILES + 15) / 16, D1 / 64);   // NT=4 -> 64 cols/group
    gemm_wmma_lds_kernel<4><<<g1, 256, 0, stream>>>(xh, wp1[0], Bl1[0], qf, INC, D1, MTILES);
    gemm_wmma_lds_kernel<4><<<g1, 256, 0, stream>>>(xh, wp1[1], Bl1[1], kf, INC, D1, MTILES);
    gemm_wmma_lds_kernel<4><<<g1, 256, 0, stream>>>(xh, wp1[2], Bl1[2], vf, INC, D1, MTILES);
    gemm_wmma_lds_kernel<4><<<g1, 256, 0, stream>>>(xh, wp1[3], Bl1[3], sf, INC, D1, MTILES);

    fill_u32_kernel<<<blocks((size_t)NN * NH, TPB), TPB, 0, stream>>>((uint32_t*)nmax, 0xFF800000u, NN * NH);
    fill_u32_kernel<<<blocks((size_t)NN * NH, TPB), TPB, 0, stream>>>((uint32_t*)den, 0u, NN * NH);
    fill_u32_kernel<<<blocks((size_t)NN * D1, TPB), TPB, 0, stream>>>((uint32_t*)agg, 0u, NN * D1);

    unsigned eb = blocks((size_t)NE * NH, TPB);
    edge_logits_kernel<<<eb, TPB, 0, stream>>>(qf, kf, src, dst, eexp, nmax);
    edge_exp_kernel<<<eb, TPB, 0, stream>>>(eexp, nmax, den, dst);
    edge_aggregate_kernel<<<eb, TPB, 0, stream>>>(eexp, den, vf, src, dst, agg);

    add_act_cast_kernel<<<blocks((size_t)NN * D1, TPB), TPB, 0, stream>>>(agg, sf, hH, NN * D1, 1);

    // ---- layer 2 ----
    gemm_wmma_lds_kernel<4><<<g1, 256, 0, stream>>>(hH, wp2[0], Bl2[0], qf, D1, D1, MTILES);
    gemm_wmma_lds_kernel<4><<<g1, 256, 0, stream>>>(hH, wp2[1], Bl2[1], kf, D1, D1, MTILES);
    gemm_wmma_lds_kernel<4><<<g1, 256, 0, stream>>>(hH, wp2[2], Bl2[2], vf, D1, D1, MTILES);
    gemm_wmma_lds_kernel<4><<<g1, 256, 0, stream>>>(hH, wp2[3], Bl2[3], sf, D1, D1, MTILES);

    fill_u32_kernel<<<blocks((size_t)NN * NH, TPB), TPB, 0, stream>>>((uint32_t*)nmax, 0xFF800000u, NN * NH);
    fill_u32_kernel<<<blocks((size_t)NN * NH, TPB), TPB, 0, stream>>>((uint32_t*)den, 0u, NN * NH);
    fill_u32_kernel<<<blocks((size_t)NN * D1, TPB), TPB, 0, stream>>>((uint32_t*)agg, 0u, NN * D1);

    edge_logits_kernel<<<eb, TPB, 0, stream>>>(qf, kf, src, dst, eexp, nmax);
    edge_exp_kernel<<<eb, TPB, 0, stream>>>(eexp, nmax, den, dst);
    edge_aggregate_kernel<<<eb, TPB, 0, stream>>>(eexp, den, vf, src, dst, agg);

    add_act_cast_kernel<<<blocks((size_t)NN * D1, TPB), TPB, 0, stream>>>(agg, sf, hH, NN * D1, 0);

    // ---- final linear + mean ----
    dim3 g2((MTILES + 15) / 16, 1);         // NT=2 -> all 32 cols in one group
    gemm_wmma_lds_kernel<2><<<g2, 256, 0, stream>>>(hH, wplin, linB, tmpo, D1, OC, MTILES);

    fill_u32_kernel<<<1, 32, 0, stream>>>((uint32_t*)out, 0u, OC);
    mean_reduce_kernel<<<64, TPB, 0, stream>>>(tmpo, out);
}
